// Graph_Memory_Attn_66529043415391
// MI455X (gfx1250) — matrix operand order repaired and checked by reading, not compiled, not run on hardware
//
#include <hip/hip_runtime.h>

// ---------------- types ----------------
typedef __attribute__((ext_vector_type(16))) __bf16 v16bf;
typedef __attribute__((ext_vector_type(8)))  __bf16 bf16x8;
typedef __attribute__((ext_vector_type(8)))  float  v8f;

#define NODEv 307
#define NPAD  320
#define SSTR  328   // LDS row stride (floats), skewed to spread banks

__device__ __forceinline__ v8f wmma_bf16(v16bf a, v16bf b, v8f c) {
  return __builtin_amdgcn_wmma_f32_16x16x32_bf16(false, a, false, b, (short)0, c, false, false);
}

// A fragment (16x32 bf16) from an fp32 row: two contiguous runs of 8
__device__ __forceinline__ v16bf load_a_f32(const float* rowp, int kbase, int lane) {
  int kb = (lane & 16) ? 16 : 0;
  v8f r0 = *(const v8f*)(rowp + kbase + kb);
  v8f r1 = *(const v8f*)(rowp + kbase + kb + 8);
  v16bf a;
  #pragma unroll
  for (int i = 0; i < 8; i++) { a[i] = (__bf16)r0[i]; a[i + 8] = (__bf16)r1[i]; }
  return a;
}
// A fragment from a bf16 row
__device__ __forceinline__ v16bf load_a_bf16(const __bf16* rowp, int kbase, int lane) {
  int kb = (lane & 16) ? 16 : 0;
  bf16x8 r0 = *(const bf16x8*)(rowp + kbase + kb);
  bf16x8 r1 = *(const bf16x8*)(rowp + kbase + kb + 8);
  v16bf a;
  #pragma unroll
  for (int i = 0; i < 8; i++) { a[i] = r0[i]; a[i + 8] = r1[i]; }
  return a;
}
// B fragment: 16 contiguous K values (fp32 source)
__device__ __forceinline__ v16bf load_b_f32(const float* p) {
  v8f r0 = *(const v8f*)p; v8f r1 = *(const v8f*)(p + 8);
  v16bf b;
  #pragma unroll
  for (int i = 0; i < 8; i++) { b[i] = (__bf16)r0[i]; b[i + 8] = (__bf16)r1[i]; }
  return b;
}

// ---------------- wave32 reductions ----------------
__device__ __forceinline__ float wave_max(float v) {
  #pragma unroll
  for (int o = 16; o > 0; o >>= 1) v = fmaxf(v, __shfl_xor(v, o, 32));
  return v;
}
__device__ __forceinline__ float wave_sum(float v) {
  #pragma unroll
  for (int o = 16; o > 0; o >>= 1) v += __shfl_xor(v, o, 32);
  return v;
}
__device__ __forceinline__ void wave_argmin(float& v, int& i) {
  #pragma unroll
  for (int o = 16; o > 0; o >>= 1) {
    float ov = __shfl_xor(v, o, 32);
    int   oi = __shfl_xor(i, o, 32);
    if (ov < v || (ov == v && oi < i)) { v = ov; i = oi; }
  }
}

// ---------------- 1) codebook softmax + entropy ----------------
__global__ void prep_kernel(const float* __restrict__ m1m, const float* __restrict__ m1r,
                            float* msm_m, float* ent_m, float* msm_r, float* ent_r) {
  int t = threadIdx.x;                       // 128 threads: 64 mem rows, 64 res rows
  const float* src = (t < 64) ? m1m : m1r;
  float* dsm = (t < 64) ? msm_m : msm_r;
  float* den = (t < 64) ? ent_m : ent_r;
  int row = t & 63;
  float v[64]; float mx = -3.0e38f;
  for (int h = 0; h < 64; h++) { v[h] = src[row * 64 + h]; mx = fmaxf(mx, v[h]); }
  float s = 0.f;
  for (int h = 0; h < 64; h++) { v[h] = __expf(v[h] - mx); s += v[h]; }
  float inv = 1.f / s; float ent = 0.f;
  for (int h = 0; h < 64; h++) {
    float p = v[h] * inv;
    dsm[row * 64 + h] = p;
    ent += p * __logf(fmaxf(p, 1e-30f));
  }
  den[row] = ent;
}

// ---------------- 2) memory module (both branches), fp32 ----------------
__global__ __launch_bounds__(256) void memory_kernel(
    const float* __restrict__ x,
    const float* __restrict__ m1m, const float* __restrict__ c1wm, const float* __restrict__ c1bm,
    const float* __restrict__ c2wm, const float* __restrict__ c2bm,
    const float* __restrict__ m1r, const float* __restrict__ c1wr, const float* __restrict__ c1br,
    const float* __restrict__ c2wr, const float* __restrict__ c2br,
    const float* __restrict__ msm_m, const float* __restrict__ ent_m,
    const float* __restrict__ msm_r, const float* __restrict__ ent_r,
    float* __restrict__ outm, float* __restrict__ outr) {
  __shared__ float buf[8][4][64];            // per-wave: x, logp, emb, sel
  int w = threadIdx.x >> 5, lane = threadIdx.x & 31;
  int row = blockIdx.x * 8 + w;              // 58944 rows exactly (fits int)
  float* xb = buf[w][0]; float* lp = buf[w][1];
  float* eb = buf[w][2]; float* sb = buf[w][3];
  int o0 = lane * 2;
  xb[o0]     = x[row * 64 + o0];
  xb[o0 + 1] = x[row * 64 + o0 + 1];
  __syncthreads();
  #pragma unroll 1
  for (int br = 0; br < 2; br++) {
    const float* m1  = br ? m1r  : m1m;
    const float* c1w = br ? c1wr : c1wm;  const float* c1b = br ? c1br : c1bm;
    const float* c2w = br ? c2wr : c2wm;  const float* c2b = br ? c2br : c2bm;
    const float* msm = br ? msm_r : msm_m; const float* ent = br ? ent_r : ent_m;
    // conv1 + relu
    float a0 = c1b[o0], a1 = c1b[o0 + 1];
    for (int h = 0; h < 64; h++) {
      float xv = xb[h];
      a0 += xv * c1w[o0 * 64 + h];
      a1 += xv * c1w[(o0 + 1) * 64 + h];
    }
    a0 = fmaxf(a0, 0.f); a1 = fmaxf(a1, 0.f);
    // log_softmax over 64 channels (2 per lane)
    float mx = wave_max(fmaxf(a0, a1));
    float se = wave_sum(__expf(a0 - mx) + __expf(a1 - mx));
    float lz = mx + __logf(se);
    lp[o0] = a0 - lz; lp[o0 + 1] = a1 - lz;
    eb[o0] = a0;      eb[o0 + 1] = a1;
    __syncthreads();
    // KL per code + argmin (64 codes, 2 per lane)
    float kl0 = ent[o0], kl1 = ent[o0 + 1];
    for (int h = 0; h < 64; h++) {
      float l = lp[h];
      kl0 -= msm[o0 * 64 + h] * l;
      kl1 -= msm[(o0 + 1) * 64 + h] * l;
    }
    float bv; int bi;
    if (kl0 <= kl1) { bv = kl0; bi = o0; } else { bv = kl1; bi = o0 + 1; }
    wave_argmin(bv, bi);
    float s0 = m1[bi * 64 + o0], s1 = m1[bi * 64 + o0 + 1];
    if (br) { s0 = a0 - s0; s1 = a1 - s1; }   // residual branch
    sb[o0] = s0; sb[o0 + 1] = s1;
    __syncthreads();
    // conv2 on concat(emb, sel)
    float r0 = c2b[o0], r1 = c2b[o0 + 1];
    for (int c = 0; c < 64; c++) {
      float e = eb[c], s = sb[c];
      r0 += e * c2w[o0 * 128 + c]       + s * c2w[o0 * 128 + 64 + c];
      r1 += e * c2w[(o0 + 1) * 128 + c] + s * c2w[(o0 + 1) * 128 + 64 + c];
    }
    float* dst = br ? outr : outm;
    dst[row * 64 + o0] = r0; dst[row * 64 + o0 + 1] = r1;
    __syncthreads();
  }
}

// ---------------- 3) zero padded key/value rows (n in [307,320)) ----------------
__global__ void pad_zero(__bf16* q_r, __bf16* k_r, __bf16* v_t) {
  int i = blockIdx.x * 256 + threadIdx.x;    // 192*8*13*64 = 1,277,952 threads exactly
  int h = i & 63; int j = i >> 6;
  int r = j % 13; int bte = j / 13;
  int n = 307 + r;
  q_r[(bte * NPAD + n) * 64 + h] = (__bf16)0.f;
  k_r[(bte * NPAD + n) * 64 + h] = (__bf16)0.f;
  v_t[(bte * 64 + h) * NPAD + n] = (__bf16)0.f;
}

// ---------------- 4) Q/K/V projection: WMMA bf16, A resident across 32 col tiles --
__global__ __launch_bounds__(128) void qkv_kernel(
    const float* __restrict__ memo, const float* __restrict__ resm,
    const float* __restrict__ qw, const float* __restrict__ qb,
    const float* __restrict__ kw, const float* __restrict__ kb_,
    const float* __restrict__ vw, const float* __restrict__ vb,
    __bf16* __restrict__ q_r, __bf16* __restrict__ k_r, __bf16* __restrict__ v_t) {
  int lane = threadIdx.x & 31, wid = threadIdx.x >> 5;
  int rt = blockIdx.x * 4 + wid;             // 3684 row tiles = 921 * 4
  int z  = blockIdx.y;                       // 0=q 1=k 2=v
  const float* A  = (z == 2) ? resm : memo;
  const float* W  = (z == 0) ? qw : (z == 1) ? kw : vw;
  const float* Bb = (z == 0) ? qb : (z == 1) ? kb_ : vb;
  int row = rt * 16 + (lane & 15);
  const float* rowp = A + row * 64;
  v16bf a0 = load_a_f32(rowp, 0, lane);
  v16bf a1 = load_a_f32(rowp, 32, lane);
  int N = lane & 15, mo = (lane & 16) ? 8 : 0;
  int ks = (lane & 16) ? 16 : 0;
  // row -> (bt, n) for this lane's 8 C elements, hoisted out of the ct loop
  int btA[8], nA[8];
  #pragma unroll
  for (int r = 0; r < 8; r++) {
    int Mg = rt * 16 + r + mo;
    btA[r] = Mg / NODEv; nA[r] = Mg - btA[r] * NODEv;
  }
  #pragma unroll 2
  for (int ct = 0; ct < 32; ct++) {
    int out = ct * 16 + N;
    v16bf b0 = load_b_f32(W + out * 64 + ks);
    v16bf b1 = load_b_f32(W + out * 64 + 32 + ks);
    v8f c = {};
    c = wmma_bf16(a0, b0, c);
    c = wmma_bf16(a1, b1, c);
    float bias = Bb[out];
    int e = out & 7, h = out >> 3;           // torch reshape: f = h*HEAD + e
    if (z == 0) {
      #pragma unroll
      for (int r = 0; r < 8; r++)
        q_r[((btA[r] * 8 + e) * NPAD + nA[r]) * 64 + h] = (__bf16)fmaxf(c[r] + bias, 0.f);
    } else if (z == 1) {
      #pragma unroll
      for (int r = 0; r < 8; r++)
        k_r[((btA[r] * 8 + e) * NPAD + nA[r]) * 64 + h] = (__bf16)fmaxf(c[r] + bias, 0.f);
    } else {
      #pragma unroll
      for (int r = 0; r < 8; r++)
        v_t[((btA[r] * 8 + e) * 64 + h) * NPAD + nA[r]] = (__bf16)fmaxf(c[r] + bias, 0.f);
    }
  }
}

// ---------------- 5) fused attention: scores WMMA -> mask -> softmax -> P@V WMMA,
//                     T-loop inside block so the time-sum stays in registers -------
__global__ __launch_bounds__(32) void attn_kernel(
    const __bf16* __restrict__ q_r, const __bf16* __restrict__ k_r,
    const __bf16* __restrict__ v_t, const float* __restrict__ stru,
    __bf16* __restrict__ pre) {
  __shared__ float    S [16 * SSTR];         // fp32 scores / probabilities
  __shared__ _Float16 SH[16 * SSTR];         // hardsigmoid(stru) mask, fp16 is plenty
  int lane = threadIdx.x;
  int bx = blockIdx.x;
  int qt = bx % 20; int e = (bx / 20) % 8; int b = bx / 160;
  for (int idx = lane; idx < 16 * NPAD; idx += 32) {
    int r = idx / NPAD, m = idx % NPAD;
    int n = qt * 16 + r; if (n > 306) n = 306;
    int mm = (m > 306) ? 306 : m;
    float v = stru[n * NODEv + mm] * (1.f / 6.f) + 0.5f;
    SH[r * SSTR + m] = (_Float16)fminf(fmaxf(v, 0.f), 1.f);
  }
  v8f O0 = {}, O1 = {}, O2 = {}, O3 = {};
  __syncthreads();
  int Mr = lane & 15;
  int ks = (lane & 16) ? 16 : 0;
  int kb = (lane & 16) ? 16 : 0;
  int mo = (lane & 16) ? 8 : 0;
  #pragma unroll 1
  for (int t = 0; t < 12; t++) {
    int bte = (b * 12 + t) * 8 + e;
    const __bf16* qbase = q_r + (bte * NPAD + qt * 16) * 64;
    const __bf16* kbase = k_r + bte * NPAD * 64;
    const __bf16* vbase = v_t + bte * 64 * NPAD;
    v16bf aq0 = load_a_bf16(qbase + Mr * 64, 0, lane);
    v16bf aq1 = load_a_bf16(qbase + Mr * 64, 32, lane);
    // scores: S = (Q K^T)/8 * hardsigmoid(stru)
    #pragma unroll 2
    for (int j = 0; j < 20; j++) {
      int key = j * 16 + (lane & 15);
      const __bf16* kp = kbase + key * 64;
      v16bf b0 = *(const v16bf*)(kp + ks);
      v16bf b1 = *(const v16bf*)(kp + 32 + ks);
      v8f c = {};
      c = wmma_bf16(aq0, b0, c);
      c = wmma_bf16(aq1, b1, c);
      int m = j * 16 + (lane & 15);
      #pragma unroll
      for (int r = 0; r < 8; r++) {
        int Mg = r + mo;
        S[Mg * SSTR + m] = c[r] * 0.125f * (float)SH[Mg * SSTR + m];
      }
    }
    __syncthreads();
    // softmax per query row over 307 valid keys; zero the pad keys
    #pragma unroll 1
    for (int r = 0; r < 16; r++) {
      float mx = -3.0e38f;
      for (int m = lane; m < NODEv; m += 32) mx = fmaxf(mx, S[r * SSTR + m]);
      mx = wave_max(mx);
      float s = 0.f;
      for (int m = lane; m < NODEv; m += 32) s += __expf(S[r * SSTR + m] - mx);
      s = wave_sum(s);
      float inv = 1.f / s;
      for (int m = lane; m < NODEv; m += 32) S[r * SSTR + m] = __expf(S[r * SSTR + m] - mx) * inv;
      int m2 = NODEv + lane;
      if (m2 < NPAD) S[r * SSTR + m2] = 0.f;
    }
    __syncthreads();
    // O += P @ V  (accumulates over t as well)
    #pragma unroll
    for (int kc = 0; kc < 10; kc++) {
      const float* sp = &S[(lane & 15) * SSTR + kc * 32];
      v8f r0 = *(const v8f*)(sp + kb);
      v8f r1 = *(const v8f*)(sp + kb + 8);
      v16bf ap;
      #pragma unroll
      for (int i = 0; i < 8; i++) { ap[i] = (__bf16)r0[i]; ap[i + 8] = (__bf16)r1[i]; }
      int keyst = kc * 32 + ks;
      O0 = wmma_bf16(ap, *(const v16bf*)(vbase + ( 0 + (lane & 15)) * NPAD + keyst), O0);
      O1 = wmma_bf16(ap, *(const v16bf*)(vbase + (16 + (lane & 15)) * NPAD + keyst), O1);
      O2 = wmma_bf16(ap, *(const v16bf*)(vbase + (32 + (lane & 15)) * NPAD + keyst), O2);
      O3 = wmma_bf16(ap, *(const v16bf*)(vbase + (48 + (lane & 15)) * NPAD + keyst), O3);
    }
    __syncthreads();
  }
  // store head-major features: pre[b][n][e*64 + h] (bf16 for dense1 WMMA)
  int N = lane & 15;
  #pragma unroll
  for (int r = 0; r < 8; r++) {
    int n = qt * 16 + r + mo;
    if (n < NODEv) {
      int base = (b * NODEv + n) * 512 + e * 64;
      pre[base +  0 + N] = (__bf16)O0[r];
      pre[base + 16 + N] = (__bf16)O1[r];
      pre[base + 32 + N] = (__bf16)O2[r];
      pre[base + 48 + N] = (__bf16)O3[r];
    }
  }
}

// ---------------- 6) dense1 WMMA + relu + transpose store -----------------------
__global__ __launch_bounds__(128) void dense_kernel(
    const __bf16* __restrict__ pre, const float* __restrict__ w,
    const float* __restrict__ bias, float* __restrict__ out) {
  int lane = threadIdx.x & 31, ct = threadIdx.x >> 5;   // 4 waves = 4 col tiles
  int rt = blockIdx.x;                                  // 307 row tiles (4912 rows exact)
  int row = rt * 16 + (lane & 15);
  const __bf16* rowp = pre + row * 512;
  int o  = ct * 16 + (lane & 15);
  int ks = (lane & 16) ? 16 : 0;
  v8f c = {};
  #pragma unroll 4
  for (int kc = 0; kc < 16; kc++) {
    v16bf a = load_a_bf16(rowp, kc * 32, lane);
    v16bf b = load_b_f32(w + o * 512 + kc * 32 + ks);
    c = wmma_bf16(a, b, c);
  }
  float bs = bias[o];
  int N = lane & 15, mo = (lane & 16) ? 8 : 0;
  #pragma unroll
  for (int r = 0; r < 8; r++) {
    int Rg = rt * 16 + r + mo;
    int bI = Rg / NODEv, n = Rg - bI * NODEv;
    float v = fmaxf(c[r] + bs, 0.f);
    out[(bI * 64 + ct * 16 + N) * NODEv + n] = v;       // [B,H,node,1]
  }
}

// ---------------- host ----------------
extern "C" void kernel_launch(void* const* d_in, const int* in_sizes, int n_in,
                              void* d_out, int out_size, void* d_ws, size_t ws_size,
                              hipStream_t stream) {
  const float* input   = (const float*)d_in[0];
  // d_in[1] topo: unused by the reference math
  const float* mem_m1  = (const float*)d_in[2];
  const float* mem_c1w = (const float*)d_in[3];
  const float* mem_c1b = (const float*)d_in[4];
  const float* mem_c2w = (const float*)d_in[5];
  const float* mem_c2b = (const float*)d_in[6];
  const float* res_m1  = (const float*)d_in[7];
  const float* res_c1w = (const float*)d_in[8];
  const float* res_c1b = (const float*)d_in[9];
  const float* res_c2w = (const float*)d_in[10];
  const float* res_c2b = (const float*)d_in[11];
  const float* q_w = (const float*)d_in[12];
  const float* q_b = (const float*)d_in[13];
  const float* k_w = (const float*)d_in[14];
  const float* k_b = (const float*)d_in[15];
  const float* v_w = (const float*)d_in[16];
  const float* v_b = (const float*)d_in[17];
  const float* stru = (const float*)d_in[18];
  const float* d1_w = (const float*)d_in[19];
  const float* d1_b = (const float*)d_in[20];

  char* ws = (char*)d_ws;
  size_t off = 0;
  auto alloc = [&](size_t bytes) -> void* {
    void* p = ws + off;
    off = (off + bytes + 255) & ~(size_t)255;
    return p;
  };
  const size_t ROWS = 58944;                  // 192 * 307
  float*  msm_m = (float*)alloc(64 * 64 * 4);
  float*  ent_m = (float*)alloc(64 * 4);
  float*  msm_r = (float*)alloc(64 * 64 * 4);
  float*  ent_r = (float*)alloc(64 * 4);
  float*  memo  = (float*)alloc(ROWS * 64 * 4);
  float*  resm  = (float*)alloc(ROWS * 64 * 4);
  __bf16* q_r = (__bf16*)alloc((size_t)192 * 8 * NPAD * 64 * 2);
  __bf16* k_r = (__bf16*)alloc((size_t)192 * 8 * NPAD * 64 * 2);
  __bf16* v_t = (__bf16*)alloc((size_t)192 * 8 * 64 * NPAD * 2);
  __bf16* pre = (__bf16*)alloc((size_t)16 * NODEv * 512 * 2);

  prep_kernel<<<1, 128, 0, stream>>>(mem_m1, res_m1, msm_m, ent_m, msm_r, ent_r);
  memory_kernel<<<7368, 256, 0, stream>>>(input,
      mem_m1, mem_c1w, mem_c1b, mem_c2w, mem_c2b,
      res_m1, res_c1w, res_c1b, res_c2w, res_c2b,
      msm_m, ent_m, msm_r, ent_r, memo, resm);
  pad_zero<<<4992, 256, 0, stream>>>(q_r, k_r, v_t);     // 192*8*13*64 / 256
  qkv_kernel<<<dim3(921, 3), 128, 0, stream>>>(memo, resm,
      q_w, q_b, k_w, k_b, v_w, v_b, q_r, k_r, v_t);
  attn_kernel<<<2560, 32, 0, stream>>>(q_r, k_r, v_t, stru, pre);
  dense_kernel<<<307, 128, 0, stream>>>(pre, d1_w, d1_b, (float*)d_out);
}
